// Qwen2MLP_withQRUN_41223096107774
// MI455X (gfx1250) — matrix operand order repaired and checked by reading, not compiled, not run on hardware
//
#include <hip/hip_runtime.h>
#include <math.h>

// ---------------------------------------------------------------------------
// Qwen2 MLP with Q-RUN layers — MI455X (gfx1250, wave32)
//
// Roofline: fp32 weights dominate (~850MB @ 23.3TB/s => ~36us floor).
// Strategy: TDM streams fp32 weight/activation tiles into LDS (async,
// TENSORcnt double-buffered, TDM applies per-row LDS padding), VALU converts
// to bf16 once, v_wmma_f32_16x16x32_bf16 does the math (~16K FLOP/inst keeps
// compute at/under the memory floor; f32 WMMA K=4 would be ~5x compute bound).
// The tiny per-channel Q-RUN MLP folds (sin/cos features tiled 3x) and fuses
// into the GEMM epilogue.
// ---------------------------------------------------------------------------

typedef __attribute__((ext_vector_type(16))) __bf16 v16bf;
typedef __attribute__((ext_vector_type(8)))  float  v8f;
typedef __attribute__((ext_vector_type(4)))  unsigned v4u;
typedef __attribute__((ext_vector_type(8)))  int    v8i;
typedef __attribute__((ext_vector_type(4)))  int    v4i;

constexpr int Mdim = 256;     // B*S
constexpr int Hdim = 5120;
constexpr int Idim = 27648;
constexpr int TM   = 128;     // M tile per workgroup (grid.y = 2)
constexpr int TN   = 64;      // N tile per workgroup
constexpr int TK   = 32;      // K step (bf16 WMMA K)
constexpr int LDA  = TK + 2;  // bf16 A-tile row stride (elements) == TDM-padded
constexpr int LDB  = TN + 2;  // bf16 B-tile row stride (elements), [k][n]
constexpr int SROW = 33;      // f32 staging row stride in dwords (TDM pad 32+1)

#if defined(__has_include)
#if __has_include(<hip/amd_detail/amd_gfx1250_TDM.h>)
#define TDM_SIX_ARGS 1
#endif
#endif

__device__ __forceinline__ unsigned short bf16one(float a) {
  unsigned ua = __builtin_bit_cast(unsigned, a);
  ua = (ua + 0x7fffu + ((ua >> 16) & 1u)) >> 16;   // RNE
  return (unsigned short)ua;
}
__device__ __forceinline__ unsigned bf16pack(float a, float b) {
  return (unsigned)bf16one(a) | ((unsigned)bf16one(b) << 16);
}
__device__ __forceinline__ unsigned lds_off_of(const void* p) {
  return (unsigned)(unsigned long long)(uintptr_t)p;  // low 32b = LDS offset
}

// --- Tensor Data Mover: async 2-D tile (global -> LDS), row padding in LDS --
// elem_code: 0=1B,1=2B,2=4B ; pad_code: pad after 2<<pad_code dwords, +1 dword
__device__ __forceinline__ void tdm_load_2d(unsigned lds_byte_off,
                                            const void* gaddr, int elem_code,
                                            int tile_x, int tile_y,
                                            int stride_elems, int pad_code,
                                            unsigned tdim0, unsigned tdim1) {
  unsigned long long ga = (unsigned long long)(uintptr_t)gaddr;
  v4u g0;
  g0.x = 1u;                                           // count=1 (valid D#)
  g0.y = lds_byte_off;                                 // lds_addr
  g0.z = (unsigned)(ga & 0xffffffffu);                 // global_addr[31:0]
  g0.w = (unsigned)((ga >> 32) & 0x01ffffffu) | (2u << 30);  // addr hi | type=2
  v8i g1;
  g1[0] = (elem_code << 16) | (1 << 20) | (pad_code << 22);  // data_size|pad_en|interval, amount=1dw
  g1[1] = (int)((tdim0 & 0xffffu) << 16);                    // tensor_dim0 lo
  g1[2] = (int)(((tdim0 >> 16) & 0xffffu) | ((tdim1 & 0xffffu) << 16));
  g1[3] = (int)(((tdim1 >> 16) & 0xffffu) | ((unsigned)tile_x << 16));  // tile_dim0
  g1[4] = tile_y & 0xffff;                                   // tile_dim1 (dim2=0)
  g1[5] = stride_elems;                                      // tensor_dim0_stride
  g1[6] = 0;
  g1[7] = 0;
  v4i z4 = {0, 0, 0, 0};
#if defined(TDM_SIX_ARGS)
  v8i z8 = {0, 0, 0, 0, 0, 0, 0, 0};
  __builtin_amdgcn_tensor_load_to_lds(g0, g1, z4, z4, z8, 0);
#else
  __builtin_amdgcn_tensor_load_to_lds(g0, g1, z4, z4, 0);
#endif
}

// A fragment (16x32 bf16): lanes 0-15 row M=lane, K {0..7,16..23};
// lanes 16-31 row M=lane-16, K {8..15,24..31}.
__device__ __forceinline__ v16bf load_frag_a(const unsigned short* Asm,
                                             int mrow, int half) {
  union { v16bf v; unsigned u[8]; } f;
  const unsigned* p = (const unsigned*)(Asm + mrow * LDA);
  const int b = half * 4;
#pragma unroll
  for (int i = 0; i < 4; ++i) f.u[i] = p[b + i];
#pragma unroll
  for (int i = 4; i < 8; ++i) f.u[i] = p[4 + b + i];
  return f.v;
}
// B fragment (32x16 bf16): lane holds column N=lane%16; lanes 0-15 K {0..7,16..23},
// lanes 16-31 K {8..15,24..31}.
__device__ __forceinline__ v16bf load_frag_b(const unsigned short* Bsm,
                                             int lane, int nbase) {
  union { v16bf v; unsigned short u[16]; } f;
  const unsigned short* p = Bsm + nbase + (lane & 15);
  const int h8 = (lane >> 4) * 8;
#pragma unroll
  for (int s = 0; s < 8; ++s) {
    f.u[s] = p[(h8 + s) * LDB];
    f.u[8 + s] = p[(16 + h8 + s) * LDB];
  }
  return f.v;
}

// ---------------------------------------------------------------------------
template <bool A_IS_BF16>
__global__ __launch_bounds__(256) void qrun_gemm_kernel(
    const void*  __restrict__ Aptr,  // [Mdim,K] f32 (or bf16 when A_IS_BF16)
    const float* __restrict__ W,     // [Ntot,K] row-major f32
    const float* __restrict__ bp, const float* __restrict__ W1,
    const float* __restrict__ b1, const float* __restrict__ W2,
    const float* __restrict__ b2,
    float* __restrict__ Out,         // [Mdim, 2*Ntot]
    int K, int Ntot) {
  // LDS layout (bytes):
  constexpr int SA_BUF  = A_IS_BF16 ? (TM * LDA * 2)   // TDM lands bf16 A-tile
                                    : (TM * SROW * 4); // f32 staging
  constexpr int SB_BUF  = TN * SROW * 4;
  constexpr int STB_OFF = 2 * SA_BUF;
  constexpr int ABF_OFF = A_IS_BF16 ? 0 : (STB_OFF + 2 * SB_BUF);
  constexpr int ABF_SZ  = A_IS_BF16 ? 0 : (TM * LDA * 2);
  constexpr int BBF_OFF = A_IS_BF16 ? (STB_OFF + 2 * SB_BUF) : (ABF_OFF + ABF_SZ);
  constexpr int CON_OFF = BBF_OFF + TK * LDB * 2;
  constexpr int SMEM_SZ = CON_OFF + 6 * 32 * 4;
  __shared__ alignas(16) char smem[SMEM_SZ];

  float*          stA  = (float*)(smem);
  float*          stB  = (float*)(smem + STB_OFF);
  unsigned short* Abf  = (unsigned short*)(smem + ABF_OFF);
  unsigned short* Bbf  = (unsigned short*)(smem + BBF_OFF);
  float* sW1s = (float*)(smem + CON_OFF);
  float* sW1c = sW1s + 32;
  float* sB1  = sW1s + 64;
  float* sW2a = sW1s + 96;
  float* sW2b = sW1s + 128;
  float* sB2  = sW1s + 160;

  const int tid  = threadIdx.x;
  const int lane = tid & 31;
  const int wave = tid >> 5;
  const int wm   = wave & 3;   // 4 wave-rows x 32 M
  const int wn   = wave >> 2;  // 2 wave-cols x 32 N
  const int half = lane >> 4;
  const int l15  = lane & 15;
  const int n0   = blockIdx.x * TN;
  const int m0   = blockIdx.y * TM;
  const unsigned smem_lds = lds_off_of(smem);

  if (tid < 32) {  // fold tiled sin/cos reuploads
    sW1s[tid] = W1[tid * 6 + 0] + W1[tid * 6 + 2] + W1[tid * 6 + 4];
    sW1c[tid] = W1[tid * 6 + 1] + W1[tid * 6 + 3] + W1[tid * 6 + 5];
    sB1[tid]  = b1[tid];
    sW2a[tid] = W2[tid];
    sW2b[tid] = W2[32 + tid];
  }
  if (tid < 2) sB2[tid] = b2[tid];

  const int iters = K / TK;

  // TDM pair for K-step s into staging buffer (s & 1). Wave-level (wave 0).
  auto issue_pair = [&](int s) {
    const int buf = s & 1;
    const int k0  = s * TK;
    if constexpr (A_IS_BF16) {
      tdm_load_2d(smem_lds + buf * SA_BUF,
                  (const unsigned short*)Aptr + (size_t)m0 * K + k0,
                  /*elem=2B*/ 1, TK, TM, K, /*pad after 16dw*/ 3,
                  (unsigned)K, (unsigned)Mdim);
    } else {
      tdm_load_2d(smem_lds + buf * SA_BUF,
                  (const float*)Aptr + (size_t)m0 * K + k0,
                  /*elem=4B*/ 2, TK, TM, K, /*pad after 32dw*/ 4,
                  (unsigned)K, (unsigned)Mdim);
    }
    tdm_load_2d(smem_lds + STB_OFF + buf * SB_BUF,
                W + (size_t)n0 * K + k0,
                2, TK, TN, K, 4, (unsigned)K, (unsigned)Ntot);
  };

  if (wave == 0) {
    issue_pair(0);
    if (iters > 1) issue_pair(1);
  }

  v8f acc[2][2];
#pragma unroll
  for (int i = 0; i < 2; ++i)
#pragma unroll
    for (int j = 0; j < 2; ++j)
      acc[i][j] = v8f{0.f, 0.f, 0.f, 0.f, 0.f, 0.f, 0.f, 0.f};

  for (int i = 0; i < iters; ++i) {
    const int buf = i & 1;
    if (wave == 0) {  // pair i complete; pair i+1 may remain in flight
      if (i + 1 < iters) __builtin_amdgcn_s_wait_tensorcnt(2);
      else               __builtin_amdgcn_s_wait_tensorcnt(0);
    }
    __syncthreads();

    // Convert staged f32 -> bf16 tiles (TDM already applied LDS row padding)
    if constexpr (!A_IS_BF16) {
      const int row = tid >> 1, hk = (tid & 1) * 16;
      const float* sr = stA + buf * (TM * SROW) + row * SROW + hk;
      unsigned* dst = (unsigned*)(Abf + row * LDA + hk);
#pragma unroll
      for (int j = 0; j < 8; ++j) dst[j] = bf16pack(sr[2 * j], sr[2 * j + 1]);
    }
    {
      const int nl = tid & 63, kq = tid >> 6;  // 64 n-cols x 4 k-octets
      const float* sr = stB + buf * (TN * SROW) + nl * SROW + kq * 8;
      unsigned short* col = Bbf + (kq * 8) * LDB + nl;
#pragma unroll
      for (int e = 0; e < 8; ++e) col[e * LDB] = bf16one(sr[e]);
    }
    __syncthreads();

    const unsigned short* Atile =
        A_IS_BF16 ? (const unsigned short*)(smem + buf * SA_BUF) : Abf;
    v16bf bfrag[2], afrag[2];
#pragma unroll
    for (int fn = 0; fn < 2; ++fn)
      bfrag[fn] = load_frag_b(Bbf, lane, wn * 32 + fn * 16);
#pragma unroll
    for (int fm = 0; fm < 2; ++fm)
      afrag[fm] = load_frag_a(Atile, wm * 32 + fm * 16 + l15, half);
#pragma unroll
    for (int fm = 0; fm < 2; ++fm)
#pragma unroll
      for (int fn = 0; fn < 2; ++fn)
        acc[fm][fn] = __builtin_amdgcn_wmma_f32_16x16x32_bf16(
            false, afrag[fm], false, bfrag[fn], (short)0, acc[fm][fn], false,
            false);
    __syncthreads();  // all LDS reads consumed -> buf is free for TDM re-fill

    if (wave == 0 && i + 2 < iters) issue_pair(i + 2);
  }
  if (wave == 0) __builtin_amdgcn_s_wait_tensorcnt(0);

  // --- fused Q-RUN epilogue: p -> (o0,o1), interleaved coalesced store -----
  const size_t ld2 = 2 * (size_t)Ntot;
#pragma unroll
  for (int fn = 0; fn < 2; ++fn) {
    const int n = n0 + wn * 32 + fn * 16 + l15;
    const float bpn = bp[n];
#pragma unroll
    for (int fm = 0; fm < 2; ++fm) {
#pragma unroll
      for (int r = 0; r < 8; ++r) {
        const int m = m0 + wm * 32 + fm * 16 + r + 8 * half;
        const float p = acc[fm][fn][r] + bpn;
        const float s = __sinf(p), c = __cosf(p);
        float o0 = sB2[0], o1 = sB2[1];
#pragma unroll 8
        for (int mh = 0; mh < 32; ++mh) {
          float hv = fmaxf(s * sW1s[mh] + c * sW1c[mh] + sB1[mh], 0.f);
          o0 += hv * sW2a[mh];
          o1 += hv * sW2b[mh];
        }
        float2 o; o.x = o0; o.y = o1;
        *reinterpret_cast<float2*>(Out + (size_t)m * ld2 + 2 * (size_t)n) = o;
      }
    }
  }
}

// h = bf16( silu(gate) * up ), vectorized x4
__global__ __launch_bounds__(256) void silu_mul_pack_kernel(
    const float* __restrict__ g, const float* __restrict__ u,
    unsigned short* __restrict__ h, int n4) {
  int i = blockIdx.x * blockDim.x + threadIdx.x;
  if (i >= n4) return;
  float4 gv = ((const float4*)g)[i];
  float4 uv = ((const float4*)u)[i];
  float a0 = gv.x / (1.f + __expf(-gv.x)) * uv.x;
  float a1 = gv.y / (1.f + __expf(-gv.y)) * uv.y;
  float a2 = gv.z / (1.f + __expf(-gv.z)) * uv.z;
  float a3 = gv.w / (1.f + __expf(-gv.w)) * uv.w;
  uint2 o; o.x = bf16pack(a0, a1); o.y = bf16pack(a2, a3);
  ((uint2*)h)[i] = o;
}

extern "C" void kernel_launch(void* const* d_in, const int* in_sizes, int n_in,
                              void* d_out, int out_size, void* d_ws,
                              size_t ws_size, hipStream_t stream) {
  const float* x   = (const float*)d_in[0];
  const float* Wg  = (const float*)d_in[1];
  const float* bg  = (const float*)d_in[2];
  const float* W1g = (const float*)d_in[3];
  const float* b1g = (const float*)d_in[4];
  const float* W2g = (const float*)d_in[5];
  const float* b2g = (const float*)d_in[6];
  const float* Wu  = (const float*)d_in[7];
  const float* bu  = (const float*)d_in[8];
  const float* W1u = (const float*)d_in[9];
  const float* b1u = (const float*)d_in[10];
  const float* W2u = (const float*)d_in[11];
  const float* b2u = (const float*)d_in[12];
  const float* Wd  = (const float*)d_in[13];
  const float* bd  = (const float*)d_in[14];
  const float* W1d = (const float*)d_in[15];
  const float* b1d = (const float*)d_in[16];
  const float* W2d = (const float*)d_in[17];
  const float* b2d = (const float*)d_in[18];

  const size_t gateBytes = (size_t)Mdim * Idim * sizeof(float);  // 28.3 MB
  char* ws = (char*)d_ws;
  float*          gate = (float*)ws;
  float*          up   = (float*)(ws + gateBytes);
  unsigned short* hbf  = (unsigned short*)(ws + 2 * gateBytes);  // bf16

  const dim3 blk(256);
  const dim3 grd1((Idim / 2) / TN, Mdim / TM);  // 216 x 2
  const dim3 grd2((Hdim / 2) / TN, Mdim / TM);  // 40 x 2

  qrun_gemm_kernel<false><<<grd1, blk, 0, stream>>>(
      x, Wg, bg, W1g, b1g, W2g, b2g, gate, Hdim, Idim / 2);
  qrun_gemm_kernel<false><<<grd1, blk, 0, stream>>>(
      x, Wu, bu, W1u, b1u, W2u, b2u, up, Hdim, Idim / 2);
  const int n4 = (Mdim * Idim) / 4;
  silu_mul_pack_kernel<<<dim3((n4 + 255) / 256), blk, 0, stream>>>(gate, up,
                                                                   hbf, n4);
  qrun_gemm_kernel<true><<<grd2, blk, 0, stream>>>(
      hbf, Wd, bd, W1d, b1d, W2d, b2d, (float*)d_out, Idim, Hdim / 2);
}
